// Keras_Graphsage_Layer_89945205112866
// MI455X (gfx1250) — matrix operand order repaired and checked by reading, not compiled, not run on hardware
//
#include <hip/hip_runtime.h>

// ---------------------------------------------------------------------------
// GraphSAGE layer, restructured:
//   out[b, j*32+q] = relu( (x3 @ W_self)[b,j,q] + C[j, b*32+q] )
//   C = adjT(bf16) @ tmpT(bf16)   -- a 4096x4096x4096 GEMM on the WMMA path
// GEMM uses double-buffered GLOBAL_LOAD_ASYNC_TO_LDS_B128 staging (ASYNCcnt)
// ---------------------------------------------------------------------------

typedef __attribute__((ext_vector_type(16))) __bf16 v16bf;
typedef __attribute__((ext_vector_type(8)))  __bf16 v8bf;
typedef __attribute__((ext_vector_type(8)))  float  v8f;

#define NN   4096
#define BB   128
#define DD   32

// Async global->LDS 16B copy, per-lane addresses. Tracked by ASYNCcnt.
__device__ __forceinline__ void async_ld_b128(unsigned lds_addr, const void* gaddr) {
    asm volatile("global_load_async_to_lds_b128 %0, %1, off"
                 :: "v"(lds_addr), "v"(gaddr)
                 : "memory");
}
__device__ __forceinline__ void wait_async0() {
    asm volatile("s_wait_asynccnt 0x0" ::: "memory");
}

// ---------------------------------------------------------------------------
// Kernel 1: adjT_bf16[j*N + i] = (bf16) adj[i*N + j]
// ---------------------------------------------------------------------------
__global__ __launch_bounds__(256)
void transpose_kernel(const float* __restrict__ adj, __bf16* __restrict__ adjT) {
    __shared__ __bf16 tile[32][33];
    const int tx = threadIdx.x & 31;
    const int ty = threadIdx.x >> 5;          // 0..7
    const int i0 = blockIdx.y * 32;
    const int j0 = blockIdx.x * 32;
    #pragma unroll
    for (int r = ty; r < 32; r += 8)
        tile[r][tx] = (__bf16)adj[(size_t)(i0 + r) * NN + j0 + tx];
    __syncthreads();
    #pragma unroll
    for (int r = ty; r < 32; r += 8)
        adjT[(size_t)(j0 + r) * NN + i0 + tx] = tile[tx][r];
}

// ---------------------------------------------------------------------------
// Kernel 2: per (b,i):  tmpT[(b*32+q)*N + i] = bf16( x3[b,i,:] . W_neigh[:,q] )
//                       out [b*N*32 + i*32+q] =      x3[b,i,:] . W_self [:,q]
// (out holds the fp32 self pre-activation; GEMM epilogue fuses relu-add.)
// ---------------------------------------------------------------------------
__global__ __launch_bounds__(256)
void prep_kernel(const float* __restrict__ x,
                 const float* __restrict__ Wn,
                 const float* __restrict__ Ws,
                 __bf16* __restrict__ tmpT,
                 float*  __restrict__ out) {
    __shared__ float sWn[DD * DD];
    __shared__ float sWs[DD * DD];
    for (int k = threadIdx.x; k < DD * DD; k += 256) {
        sWn[k] = Wn[k];
        sWs[k] = Ws[k];
    }
    __syncthreads();

    const int gid = blockIdx.x * 256 + threadIdx.x;  // over B*N
    const int b = gid >> 12;
    const int i = gid & (NN - 1);
    const float* xr = x + (size_t)b * (NN * DD) + (size_t)i * DD;

    float xv[DD];
    #pragma unroll
    for (int p = 0; p < DD; p += 4) {
        const float4 v = *(const float4*)(xr + p);
        xv[p + 0] = v.x; xv[p + 1] = v.y; xv[p + 2] = v.z; xv[p + 3] = v.w;
    }

    const size_t selfBase = (size_t)b * (NN * DD) + (size_t)i * DD;
    #pragma unroll 4
    for (int q = 0; q < DD; ++q) {
        float accN = 0.0f, accS = 0.0f;
        #pragma unroll
        for (int p = 0; p < DD; ++p) {
            accN = fmaf(xv[p], sWn[p * DD + q], accN);
            accS = fmaf(xv[p], sWs[p * DD + q], accS);
        }
        tmpT[(size_t)(b * DD + q) * NN + i] = (__bf16)accN;
        out[selfBase + q] = accS;
    }
}

// ---------------------------------------------------------------------------
// Kernel 3: C[j, n] = sum_i adjT[j,i] * tmpT[n,i], n = b*32+q
//   out[b*N*32 + j*32 + q] = relu(out[...] + C[j,n])
// WG tile 128x128, BK=32, 8 waves (2 in M x 4 in N), wave tile 64x32
//   => 4x2 accumulators, 8x v_wmma_f32_16x16x32_bf16 per K-step, 128 steps.
// Staging: double-buffered async global->LDS (4x b128 issues/thread/step).
// ---------------------------------------------------------------------------
__global__ __launch_bounds__(256)
void gemm_kernel(const __bf16* __restrict__ A,    // adjT  [4096][4096] row-major (K contig)
                 const __bf16* __restrict__ Bm,   // tmpT  [4096][4096] row-major (K contig)
                 float* __restrict__ out) {
    constexpr int TM = 128, TN = 128, TK = 32;
    constexpr int LDK = TK + 8;                       // pad: 80B row stride
    constexpr unsigned BUFB = (unsigned)(TM * LDK * 2); // 10240 B per buffer
    __shared__ __bf16 sA[2][TM * LDK];
    __shared__ __bf16 sB[2][TN * LDK];

    const int tid  = threadIdx.x;
    const int lane = tid & 31;
    const int wave = tid >> 5;
    const int wm   = wave >> 2;                   // 0..1
    const int wn   = wave & 3;                    // 0..3
    const int half = lane >> 4;                   // 0/1
    const int l15  = lane & 15;
    const int m0   = blockIdx.y * TM;
    const int n0   = blockIdx.x * TN;

    // ---- per-thread staging addresses (2 x 16B chunks per operand) ----
    const int srow0 = tid >> 2;                   // 0..63
    const int scc   = (tid & 3) * 8;              // bf16 col offset (16B units)
    const int srow1 = srow0 + 64;                 // 64..127
    const __bf16* gA0 = A  + (size_t)(m0 + srow0) * NN + scc;
    const __bf16* gA1 = A  + (size_t)(m0 + srow1) * NN + scc;
    const __bf16* gB0 = Bm + (size_t)(n0 + srow0) * NN + scc;
    const __bf16* gB1 = Bm + (size_t)(n0 + srow1) * NN + scc;
    // LDS aperture lives in addr[63:32]; addr[31:0] is the LDS byte offset.
    const unsigned lA0 = (unsigned)(uintptr_t)&sA[0][srow0 * LDK + scc];
    const unsigned lA1 = (unsigned)(uintptr_t)&sA[0][srow1 * LDK + scc];
    const unsigned lB0 = (unsigned)(uintptr_t)&sB[0][srow0 * LDK + scc];
    const unsigned lB1 = (unsigned)(uintptr_t)&sB[0][srow1 * LDK + scc];

    const v8f zacc = {0.f, 0.f, 0.f, 0.f, 0.f, 0.f, 0.f, 0.f};
    v8f acc[4][2];
    #pragma unroll
    for (int mi = 0; mi < 4; ++mi)
        #pragma unroll
        for (int ni = 0; ni < 2; ++ni)
            acc[mi][ni] = zacc;

    // ---- prologue: stage tile k0=0 into buffer 0 ----
    async_ld_b128(lA0, gA0);
    async_ld_b128(lA1, gA1);
    async_ld_b128(lB0, gB0);
    async_ld_b128(lB1, gB1);
    wait_async0();
    __syncthreads();

    int buf = 0;
    for (int k0 = 0; k0 < NN; k0 += TK, buf ^= 1) {
        // issue async staging of next tile into the other buffer
        if (k0 + TK < NN) {
            const unsigned bo = buf ? 0u : BUFB;  // destination = buf^1
            const size_t  ko = (size_t)(k0 + TK);
            async_ld_b128(lA0 + bo, gA0 + ko);
            async_ld_b128(lA1 + bo, gA1 + ko);
            async_ld_b128(lB0 + bo, gB0 + ko);
            async_ld_b128(lB1 + bo, gB1 + ko);
        }

        const __bf16* pAt = sA[buf];
        const __bf16* pBt = sB[buf];

        // A fragments: 16x32 bf16; lanes0-15 hold K0-7/K16-23, lanes16-31 K8-15/K24-31
        v16bf af[4];
        #pragma unroll
        for (int mi = 0; mi < 4; ++mi) {
            const __bf16* p = pAt + (wm * 64 + mi * 16 + l15) * LDK + half * 8;
            const v8bf lo = *(const v8bf*)(p);
            const v8bf hi = *(const v8bf*)(p + 16);
            af[mi] = __builtin_shufflevector(lo, hi, 0,1,2,3,4,5,6,7,8,9,10,11,12,13,14,15);
        }
        // B fragments: 32x16 bf16; lanes0-15 hold K0-7/K16-23, lanes16-31 K8-15/K24-31
        v16bf bfrag[2];
        #pragma unroll
        for (int ni = 0; ni < 2; ++ni) {
            const __bf16* p = pBt + (wn * 32 + ni * 16 + l15) * LDK + half * 8;
            const v8bf lo = *(const v8bf*)(p);
            const v8bf hi = *(const v8bf*)(p + 16);
            bfrag[ni] = __builtin_shufflevector(lo, hi, 0,1,2,3,4,5,6,7,8,9,10,11,12,13,14,15);
        }

        #pragma unroll
        for (int mi = 0; mi < 4; ++mi)
            #pragma unroll
            for (int ni = 0; ni < 2; ++ni)
                acc[mi][ni] = __builtin_amdgcn_wmma_f32_16x16x32_bf16(
                    false, af[mi], false, bfrag[ni],
                    (short)0, acc[mi][ni], false, false);

        // my async writes to buf^1 have landed; barrier publishes them and
        // retires everyone's reads of buf before it is re-staged next round
        wait_async0();
        __syncthreads();
    }

    // Epilogue: C layout VGPR v, lane l -> M = v + 8*(l>>4), Ncol = l&15
    #pragma unroll
    for (int mi = 0; mi < 4; ++mi) {
        #pragma unroll
        for (int ni = 0; ni < 2; ++ni) {
            const int n  = n0 + wn * 32 + ni * 16 + l15;
            const int b  = n >> 5;
            const int q  = n & 31;
            const int mr = m0 + wm * 64 + mi * 16 + half * 8;
            const size_t base = (size_t)b * (NN * DD) + (size_t)mr * DD + q;
            #pragma unroll
            for (int v = 0; v < 8; ++v) {
                const size_t idx = base + (size_t)v * DD;
                const float s = out[idx];               // fp32 self pre-activation
                out[idx] = fmaxf(s + acc[mi][ni][v], 0.0f);
            }
        }
    }
}

// ---------------------------------------------------------------------------
extern "C" void kernel_launch(void* const* d_in, const int* in_sizes, int n_in,
                              void* d_out, int out_size, void* d_ws, size_t ws_size,
                              hipStream_t stream) {
    (void)in_sizes; (void)n_in; (void)out_size; (void)ws_size;
    const float* x   = (const float*)d_in[0];   // [B, N*32] fp32
    const float* adj = (const float*)d_in[1];   // [N, N]    fp32
    const float* Wn  = (const float*)d_in[2];   // [32, 32]  fp32
    const float* Ws  = (const float*)d_in[3];   // [32, 32]  fp32
    float* out = (float*)d_out;                 // [B, N*32] fp32

    __bf16* adjT = (__bf16*)d_ws;                       // 4096*4096 bf16 = 33.5 MB
    __bf16* tmpT = adjT + (size_t)NN * NN;              // 4096*4096 bf16 = 33.5 MB

    transpose_kernel<<<dim3(NN / 32, NN / 32), 256, 0, stream>>>(adj, adjT);
    prep_kernel<<<dim3((BB * NN) / 256), 256, 0, stream>>>(x, Wn, Ws, tmpT, out);
    gemm_kernel<<<dim3(NN / 128, NN / 128), 256, 0, stream>>>(adjT, tmpT, out);
}
